// PruningNetwork_66340064854369
// MI455X (gfx1250) — matrix operand order repaired and checked by reading, not compiled, not run on hardware
//
#include <hip/hip_runtime.h>
#include <hip/hip_bf16.h>

typedef __attribute__((ext_vector_type(16))) _Float16 v16h;
typedef __attribute__((ext_vector_type(8)))  float    v8f;
typedef __attribute__((ext_vector_type(8)))  unsigned v8u;
typedef __attribute__((ext_vector_type(4)))  unsigned v4u;

__device__ __forceinline__ unsigned packh(_Float16 a, _Float16 b) {
  union { _Float16 h[2]; unsigned u; } x;
  x.h[0] = a; x.h[1] = b;
  return x.u;
}
__device__ __forceinline__ float unpackh(unsigned w, int hi) {
  union { _Float16 h[2]; unsigned u; } x;
  x.u = w;
  return (float)x.h[hi];
}

// ---------------------------------------------------------------------------
// Weight prep: fold BN/affine into f16 weights (K zero-padded) + f32 bias.
// mode 0: conv (W*=g, bias=b); 1: fc (W*=g, bias=b*g+be); 2: plain.
// ---------------------------------------------------------------------------
__global__ void prep_weights(const float* __restrict__ W, const float* __restrict__ g,
                             const float* __restrict__ b, const float* __restrict__ be,
                             _Float16* __restrict__ Wh, float* __restrict__ bias,
                             int Cout, int Cin, int Kpad, int mode) {
  int idx = blockIdx.x * blockDim.x + threadIdx.x;
  if (idx < Cout * Kpad) {
    int co = idx / Kpad, k = idx - co * Kpad;
    float s = (mode == 2) ? 1.f : g[co];
    Wh[idx] = (_Float16)((k < Cin) ? W[(size_t)co * Cin + k] * s : 0.f);
  }
  if (idx < Cout && bias != nullptr) {
    bias[idx] = (mode == 1) ? (b[idx] * g[idx] + be[idx]) : b[idx];
  }
}

// ---------------------------------------------------------------------------
// Farthest point sampling: one block per cloud, LDS dists + argmax reduction.
// ---------------------------------------------------------------------------
__global__ __launch_bounds__(256)
void fps_kernel(const float* __restrict__ xyz, int n, int npoint,
                int* __restrict__ fidx, float* __restrict__ new_xyz) {
  __shared__ float dists[2048];
  __shared__ float rval[256];
  __shared__ int   ridx[256];
  __shared__ int   s_last;
  const float* X = xyz + (size_t)blockIdx.x * n * 3;
  int t = threadIdx.x;
  for (int i = t; i < n; i += 256) dists[i] = 1e10f;
  if (t == 0) { s_last = 0; fidx[(size_t)blockIdx.x * npoint] = 0; }
  __syncthreads();
  for (int it = 1; it < npoint; ++it) {
    int last = s_last;
    float lx = X[3 * last], ly = X[3 * last + 1], lz = X[3 * last + 2];
    float bv = -1.f; int bi = 0;
    for (int i = t; i < n; i += 256) {
      float dx = X[3 * i] - lx, dy = X[3 * i + 1] - ly, dz = X[3 * i + 2] - lz;
      float nd = fminf(dists[i], dx * dx + dy * dy + dz * dz);
      dists[i] = nd;
      if (nd > bv) { bv = nd; bi = i; }
    }
    rval[t] = bv; ridx[t] = bi;
    __syncthreads();
    for (int s = 128; s > 0; s >>= 1) {
      if (t < s) {
        bool take = (rval[t + s] > rval[t]) ||
                    (rval[t + s] == rval[t] && ridx[t + s] < ridx[t]);
        if (take) { rval[t] = rval[t + s]; ridx[t] = ridx[t + s]; }
      }
      __syncthreads();
    }
    if (t == 0) { s_last = ridx[0]; fidx[(size_t)blockIdx.x * npoint + it] = ridx[0]; }
    __syncthreads();
  }
  for (int i = t; i < npoint; i += 256) {
    int p = fidx[(size_t)blockIdx.x * npoint + i];
    new_xyz[((size_t)blockIdx.x * npoint + i) * 3 + 0] = X[3 * p];
    new_xyz[((size_t)blockIdx.x * npoint + i) * 3 + 1] = X[3 * p + 1];
    new_xyz[((size_t)blockIdx.x * npoint + i) * 3 + 2] = X[3 * p + 2];
  }
}

// ---------------------------------------------------------------------------
// Ball query: one wave32 per center; ballot/popcount first-K ascending select.
// ---------------------------------------------------------------------------
__global__ __launch_bounds__(256)
void ball_query_kernel(const float* __restrict__ xyz, const float* __restrict__ centers,
                       int* __restrict__ gidx, int B, int n, int S, int K, float r2) {
  int gwid = (blockIdx.x * blockDim.x + threadIdx.x) >> 5;
  int lane = threadIdx.x & 31;
  if (gwid >= B * S) return;
  int bb = gwid / S, s = gwid - bb * S;
  const float* X = xyz + (size_t)bb * n * 3;
  const float* c = centers + ((size_t)bb * S + s) * 3;
  float cx = c[0], cy = c[1], cz = c[2];
  int* out = gidx + ((size_t)bb * S + s) * K;
  int count = 0, first = 0;
  bool haveFirst = false;
  for (int base = 0; base < n && count < K; base += 32) {
    int i = base + lane;
    bool hit = false;
    if (i < n) {
      float dx = X[3 * i] - cx, dy = X[3 * i + 1] - cy, dz = X[3 * i + 2] - cz;
      hit = (dx * dx + dy * dy + dz * dz) <= r2;
    }
    unsigned m = (unsigned)__ballot(hit);
    int pos = count + __popc(m & ((1u << lane) - 1u));
    if (hit && pos < K) out[pos] = i;
    if (!haveFirst && m != 0u) { first = base + __ffs(m) - 1; haveFirst = true; }
    count += __popc(m);
  }
  if (count > K) count = K;
  for (int j = count + lane; j < K; j += 32) out[j] = first;
}

// ---------------------------------------------------------------------------
// Grouping kernels: emit K-pair-packed u32 rows [Kpad/2][M]
// (word = {f16 ch=2p (low), f16 ch=2p+1 (high)}) matching the B-fragment.
// ---------------------------------------------------------------------------
__global__ void group_sa1(const float* __restrict__ xyz, const float* __restrict__ feats,
                          const float* __restrict__ nxyz, const int* __restrict__ gidx,
                          unsigned* __restrict__ G, int b0, int CH) {
  const int S = 128, K = 64, N = 2048, KPH = 16;  // Kpad=32 -> 16 pair rows
  int M = S * K;
  int tid = blockIdx.x * blockDim.x + threadIdx.x;
  if (tid >= CH * M) return;
  int cb = tid / M, m = tid - cb * M;
  int b = b0 + cb, s = m / K, k = m - s * K;
  int pi = gidx[((size_t)b * S + s) * K + k];
  const float* X = xyz + (size_t)b * N * 3;
  const float* c = nxyz + ((size_t)b * S + s) * 3;
  const float* F = feats + (size_t)b * 4 * N;
  float rx = X[3 * pi + 0] - c[0], ry = X[3 * pi + 1] - c[1], rz = X[3 * pi + 2] - c[2];
  float f0 = F[0 * N + pi], f1 = F[1 * N + pi], f2 = F[2 * N + pi], f3 = F[3 * N + pi];
  unsigned* g = G + (size_t)cb * KPH * M;
  g[0 * M + m] = packh((_Float16)rx, (_Float16)ry);
  g[1 * M + m] = packh((_Float16)rz, (_Float16)f0);
  g[2 * M + m] = packh((_Float16)f1, (_Float16)f2);
  g[3 * M + m] = packh((_Float16)f3, (_Float16)0.f);
  for (int p = 4; p < KPH; ++p) g[(size_t)p * M + m] = 0u;
}

__global__ void group_sa2(const float* __restrict__ nxyz1, const _Float16* __restrict__ feat1,
                          const float* __restrict__ nxyz2, const int* __restrict__ gidx2,
                          unsigned* __restrict__ G, int b0, int CH) {
  const int S = 32, K = 128, N = 128, C = 128, KPH = 80;  // Kpad=160
  int M = S * K;
  int tid = blockIdx.x * blockDim.x + threadIdx.x;
  if (tid >= CH * M) return;
  int cb = tid / M, m = tid - cb * M;
  int b = b0 + cb, s = m / K, k = m - s * K;
  int pi = gidx2[((size_t)b * S + s) * K + k];
  const float* P = nxyz1 + ((size_t)b * N + pi) * 3;
  const float* c = nxyz2 + ((size_t)b * S + s) * 3;
  const _Float16* F = feat1 + (size_t)b * C * N + pi;
  float rel[3] = {P[0] - c[0], P[1] - c[1], P[2] - c[2]};
  unsigned* g = G + (size_t)cb * KPH * M;
  for (int p = 0; p < KPH; ++p) {
    int c0 = 2 * p, c1 = 2 * p + 1;
    _Float16 h0 = (c0 < 3) ? (_Float16)rel[c0] : ((c0 - 3 < C) ? F[(size_t)(c0 - 3) * N] : (_Float16)0.f);
    _Float16 h1 = (c1 < 3) ? (_Float16)rel[c1] : ((c1 - 3 < C) ? F[(size_t)(c1 - 3) * N] : (_Float16)0.f);
    g[(size_t)p * M + m] = packh(h0, h1);
  }
}

__global__ void group_sa3(const float* __restrict__ nxyz2, const _Float16* __restrict__ feat2,
                          unsigned* __restrict__ G) {
  const int n = 32, C = 256, KPH = 144, M = 32;  // Kpad=288
  int tid = blockIdx.x * blockDim.x + threadIdx.x;
  if (tid >= 64 * M) return;
  int b = tid / M, m = tid - b * M;
  const float* P = nxyz2 + ((size_t)b * n + m) * 3;
  const _Float16* F = feat2 + (size_t)b * C * n + m;
  float ab[3] = {P[0], P[1], P[2]};
  unsigned* g = G + (size_t)b * KPH * M;
  for (int p = 0; p < KPH; ++p) {
    int c0 = 2 * p, c1 = 2 * p + 1;
    _Float16 h0 = (c0 < 3) ? (_Float16)ab[c0] : ((c0 - 3 < C) ? F[(size_t)(c0 - 3) * n] : (_Float16)0.f);
    _Float16 h1 = (c1 < 3) ? (_Float16)ab[c1] : ((c1 - 3 < C) ? F[(size_t)(c1 - 3) * n] : (_Float16)0.f);
    g[(size_t)p * M + m] = packh(h0, h1);
  }
}

// ---------------------------------------------------------------------------
// WMMA GEMM + bias + optional ReLU on K-pair-packed activations.
//   Out[co,m] = act( sum_k Wh[co,k]*In[k,m] + bias[co] )
// M is compile-time: all B-fragment loads fold row offsets (v*M*4 bytes) into
// the 24-bit instruction offset off one base address; per-chunk advance is a
// single pointer add. One wave owns a 16x32 tile: A (2 x b128) reused by two
// back-to-back WMMAs, 16 offset-folded b32 B-loads, packed b32 stores.
// ---------------------------------------------------------------------------
template <int M>
__global__ __launch_bounds__(256)
void wmma_gemm_relu(const _Float16* __restrict__ Wh, const unsigned* __restrict__ InP,
                    const float* __restrict__ bias, unsigned* __restrict__ OutP,
                    int Cout, int Kpad,
                    long long inStrideU, long long outStrideU, int doRelu) {
  constexpr int tilesM2 = M >> 5;
  int tilesC = Cout >> 4;
  int w = blockIdx.x * (blockDim.x >> 5) + (threadIdx.x >> 5);
  if (w >= tilesM2 * tilesC) return;
  int mt = w % tilesM2, ct = w / tilesM2;
  int lane = threadIdx.x & 31, half = lane >> 4, lm = lane & 15;
  const unsigned* in = InP + (size_t)blockIdx.z * inStrideU;
  unsigned* out = OutP + (size_t)blockIdx.z * outStrideU;
  int m0 = mt << 5, co0 = ct << 4;
  int col0 = m0 + lm;

  v8f acc0 = {}, acc1 = {};
  const _Float16* wrow = Wh + (size_t)(co0 + lm) * Kpad + 16 * half;
  const unsigned* bptr = in + (size_t)(8 * half) * M + col0;  // row v=0 of chunk 0
  for (int k0 = 0; k0 < Kpad; k0 += 32) {
    // A 16x32: lane's words are kb={0,2,4,6}+16h and {8,10,12,14}+16h
    v4u alo = *(const v4u*)(wrow);
    v4u ahi = *(const v4u*)(wrow + 8);
    v8u au;
    au[0] = alo[0]; au[1] = alo[1]; au[2] = alo[2]; au[3] = alo[3];
    au[4] = ahi[0]; au[5] = ahi[1]; au[6] = ahi[2]; au[7] = ahi[3];
    // B 32x16: VGPR v = pair-row (k0/2 + v + 8h); offsets are compile-time
    v8u b0u, b1u;
#pragma unroll
    for (int v = 0; v < 8; ++v) {
      b0u[v] = bptr[v * M];
      b1u[v] = bptr[v * M + 16];
    }
    wrow += 32;
    bptr += 16 * M;
    v16h a = __builtin_bit_cast(v16h, au);
    acc0 = __builtin_amdgcn_wmma_f32_16x16x32_f16(false, a, false,
        __builtin_bit_cast(v16h, b0u), (short)0, acc0, false, false);
    acc1 = __builtin_amdgcn_wmma_f32_16x16x32_f16(false, a, false,
        __builtin_bit_cast(v16h, b1u), (short)0, acc1, false, false);
  }
  // D 16x16 f32: lane holds co = co0+8h+v, v=0..7 -> 4 packed channel pairs
#pragma unroll
  for (int j = 0; j < 4; ++j) {
    int co = co0 + 8 * half + 2 * j;
    float bz0 = bias[co], bz1 = bias[co + 1];
    float x0 = acc0[2 * j] + bz0, x1 = acc0[2 * j + 1] + bz1;
    float y0 = acc1[2 * j] + bz0, y1 = acc1[2 * j + 1] + bz1;
    if (doRelu) {
      x0 = fmaxf(x0, 0.f); x1 = fmaxf(x1, 0.f);
      y0 = fmaxf(y0, 0.f); y1 = fmaxf(y1, 0.f);
    }
    size_t r = (size_t)(co >> 1) * M + col0;
    out[r]      = packh((_Float16)x0, (_Float16)x1);
    out[r + 16] = packh((_Float16)y0, (_Float16)y1);
  }
}

// ---------------------------------------------------------------------------
// Max pool over K, packed input -> plain f16 [C][S] output (per out-batch obs).
// ---------------------------------------------------------------------------
__global__ void pool_max(const unsigned* __restrict__ HP, _Float16* __restrict__ out,
                         int C, int S, int K, long long obs, int ocs) {
  int tid = blockIdx.x * blockDim.x + threadIdx.x;
  if (tid >= C * S) return;
  int c = tid / S, s = tid - c * S;
  int M = S * K;
  const _Float16* Hh = (const _Float16*)(HP + (size_t)blockIdx.z * (C >> 1) * M);
  const _Float16* base = Hh + (size_t)(c >> 1) * 2 * M + (c & 1);
  float m = -1e30f;
  for (int k = 0; k < K; ++k) {
    float v = (float)base[(size_t)(s * K + k) * 2];
    m = v > m ? v : m;
  }
  out[(size_t)blockIdx.z * obs + (size_t)c * ocs + s] = (_Float16)m;
}

// SA3 pool: packed [512][32] per batch -> packed channel-major xT[cp][64].
__global__ void pool3_packed(const unsigned* __restrict__ HP, unsigned* __restrict__ xTp) {
  int cp = blockIdx.x * blockDim.x + threadIdx.x;
  if (cp >= 512) return;
  const _Float16* base = (const _Float16*)(HP + (size_t)blockIdx.z * 512 * 32) + (size_t)cp * 64;
  float m0 = -1e30f, m1 = -1e30f;
  for (int k = 0; k < 32; ++k) {
    float v0 = (float)base[2 * k], v1 = (float)base[2 * k + 1];
    m0 = v0 > m0 ? v0 : m0;
    m1 = v1 > m1 ? v1 : m1;
  }
  xTp[(size_t)cp * 64 + blockIdx.z] = packh((_Float16)m0, (_Float16)m1);
}

// ---------------------------------------------------------------------------
// Final 1024 -> 1 dot + sigmoid over packed h [512][64]. One block per batch.
// ---------------------------------------------------------------------------
__global__ __launch_bounds__(256)
void fc3_sigmoid(const unsigned* __restrict__ hP, const _Float16* __restrict__ w3,
                 const float* __restrict__ b3, float* __restrict__ out) {
  __shared__ float red[256];
  int b = blockIdx.x, t = threadIdx.x;
  float s = 0.f;
  for (int kp = t; kp < 512; kp += 256) {
    unsigned wd = hP[(size_t)kp * 64 + b];
    s += unpackh(wd, 0) * (float)w3[2 * kp] + unpackh(wd, 1) * (float)w3[2 * kp + 1];
  }
  red[t] = s;
  __syncthreads();
  for (int st = 128; st > 0; st >>= 1) {
    if (t < st) red[t] += red[t + st];
    __syncthreads();
  }
  if (t == 0) {
    float z = red[0] + b3[0];
    out[b] = 1.f / (1.f + __expf(-z));
  }
}

// ---------------------------------------------------------------------------
extern "C" void kernel_launch(void* const* d_in, const int* in_sizes, int n_in,
                              void* d_out, int out_size, void* d_ws, size_t ws_size,
                              hipStream_t stream) {
  (void)in_sizes; (void)n_in; (void)out_size; (void)ws_size;
  const int B = 64, N = 2048;
  const float* pc  = (const float*)d_in[0];
  const float* pcf = (const float*)d_in[1];

  char* ws = (char*)d_ws;
  size_t off = 0;
  auto alloc = [&](size_t bytes) -> char* {
    off = (off + 255) & ~(size_t)255;
    char* p = ws + off;
    off += bytes;
    return p;
  };

  // conv layers: {Cout, Cin, Kpad, W idx, g idx, b idx}
  const int conv[9][6] = {
      {64,   7,   32,  2,  3,  4}, {64,  64,  64,  5,  6,  7}, {128, 64,  64,  8,  9, 10},
      {128, 131, 160, 11, 12, 13}, {128, 128, 128, 14, 15, 16}, {256, 128, 128, 17, 18, 19},
      {256, 259, 288, 20, 21, 22}, {512, 256, 256, 23, 24, 25}, {1024, 512, 512, 26, 27, 28}};
  _Float16* Whc[9]; float* biasc[9];
  for (int l = 0; l < 9; ++l) {
    Whc[l]   = (_Float16*)alloc((size_t)conv[l][0] * conv[l][2] * sizeof(_Float16));
    biasc[l] = (float*)alloc((size_t)conv[l][0] * sizeof(float));
  }
  _Float16* WhFc1 = (_Float16*)alloc((size_t)1024 * 1024 * sizeof(_Float16));
  float*    bFc1  = (float*)alloc(1024 * sizeof(float));
  _Float16* WhFc2 = (_Float16*)alloc((size_t)1024 * 1024 * sizeof(_Float16));
  float*    bFc2  = (float*)alloc(1024 * sizeof(float));
  _Float16* Wh3   = (_Float16*)alloc(1024 * sizeof(_Float16));

  int*   fidx1 = (int*)alloc((size_t)B * 128 * sizeof(int));
  float* nxyz1 = (float*)alloc((size_t)B * 128 * 3 * sizeof(float));
  int*   gidx1 = (int*)alloc((size_t)B * 128 * 64 * sizeof(int));
  int*   fidx2 = (int*)alloc((size_t)B * 32 * sizeof(int));
  float* nxyz2 = (float*)alloc((size_t)B * 32 * 3 * sizeof(float));
  int*   gidx2 = (int*)alloc((size_t)B * 32 * 128 * sizeof(int));

  _Float16* feat1 = (_Float16*)alloc((size_t)B * 128 * 128 * sizeof(_Float16));
  _Float16* feat2 = (_Float16*)alloc((size_t)B * 256 * 32 * sizeof(_Float16));
  unsigned* xTp   = (unsigned*)alloc((size_t)512 * 64 * sizeof(unsigned));

  const int CH = 8;  // batch chunk for SA1/SA2
  unsigned* Gbuf = (unsigned*)alloc((size_t)CH * 80 * 4096 * sizeof(unsigned));
  unsigned* Ha   = (unsigned*)alloc((size_t)2097152 * sizeof(unsigned));
  unsigned* Hb   = (unsigned*)alloc((size_t)2097152 * sizeof(unsigned));
  unsigned* Hc   = (unsigned*)alloc((size_t)4194304 * sizeof(unsigned));

  // ---- weight prep ----
  for (int l = 0; l < 9; ++l) {
    int total = conv[l][0] * conv[l][2];
    prep_weights<<<(total + 255) / 256, 256, 0, stream>>>(
        (const float*)d_in[conv[l][3]], (const float*)d_in[conv[l][4]],
        (const float*)d_in[conv[l][5]], nullptr, Whc[l], biasc[l],
        conv[l][0], conv[l][1], conv[l][2], 0);
  }
  prep_weights<<<(1024 * 1024 + 255) / 256, 256, 0, stream>>>(
      (const float*)d_in[29], (const float*)d_in[31], (const float*)d_in[30],
      (const float*)d_in[32], WhFc1, bFc1, 1024, 1024, 1024, 1);
  prep_weights<<<(1024 * 1024 + 255) / 256, 256, 0, stream>>>(
      (const float*)d_in[33], (const float*)d_in[35], (const float*)d_in[34],
      (const float*)d_in[36], WhFc2, bFc2, 1024, 1024, 1024, 1);
  prep_weights<<<(1024 + 255) / 256, 256, 0, stream>>>(
      (const float*)d_in[37], nullptr, nullptr, nullptr, Wh3, nullptr, 1, 1024, 1024, 2);

  auto gemm = [&](const _Float16* Wm, const unsigned* In, const float* bias, unsigned* Out,
                  int Cout, int Kpad, int M, int nb, int relu) {
    int tiles = (M >> 5) * (Cout >> 4);
    dim3 grid((tiles + 7) / 8, 1, nb);
    long long inS = (long long)(Kpad >> 1) * M;
    long long outS = (long long)(Cout >> 1) * M;
    switch (M) {
      case 8192:
        wmma_gemm_relu<8192><<<grid, 256, 0, stream>>>(Wm, In, bias, Out, Cout, Kpad, inS, outS, relu);
        break;
      case 4096:
        wmma_gemm_relu<4096><<<grid, 256, 0, stream>>>(Wm, In, bias, Out, Cout, Kpad, inS, outS, relu);
        break;
      case 64:
        wmma_gemm_relu<64><<<grid, 256, 0, stream>>>(Wm, In, bias, Out, Cout, Kpad, inS, outS, relu);
        break;
      default:
        wmma_gemm_relu<32><<<grid, 256, 0, stream>>>(Wm, In, bias, Out, Cout, Kpad, inS, outS, relu);
        break;
    }
  };

  // ---- SA1 ----
  fps_kernel<<<B, 256, 0, stream>>>(pc, N, 128, fidx1, nxyz1);
  ball_query_kernel<<<(B * 128 * 32 + 255) / 256, 256, 0, stream>>>(
      pc, nxyz1, gidx1, B, N, 128, 64, 0.025f * 0.025f);
  for (int b0 = 0; b0 < B; b0 += CH) {
    const int M1 = 128 * 64;
    group_sa1<<<(CH * M1 + 255) / 256, 256, 0, stream>>>(pc, pcf, nxyz1, gidx1, Gbuf, b0, CH);
    gemm(Whc[0], Gbuf, biasc[0], Ha, 64, 32, M1, CH, 1);
    gemm(Whc[1], Ha,   biasc[1], Hb, 64, 64, M1, CH, 1);
    gemm(Whc[2], Hb,   biasc[2], Hc, 128, 64, M1, CH, 1);
    pool_max<<<dim3((128 * 128 + 255) / 256, 1, CH), 256, 0, stream>>>(
        Hc, feat1 + (size_t)b0 * 128 * 128, 128, 128, 64, (long long)128 * 128, 128);
  }

  // ---- SA2 ----
  fps_kernel<<<B, 256, 0, stream>>>(nxyz1, 128, 32, fidx2, nxyz2);
  ball_query_kernel<<<(B * 32 * 32 + 255) / 256, 256, 0, stream>>>(
      nxyz1, nxyz2, gidx2, B, 128, 32, 128, 0.05f * 0.05f);
  for (int b0 = 0; b0 < B; b0 += CH) {
    const int M2 = 32 * 128;
    group_sa2<<<(CH * M2 + 255) / 256, 256, 0, stream>>>(nxyz1, feat1, nxyz2, gidx2, Gbuf, b0, CH);
    gemm(Whc[3], Gbuf, biasc[3], Ha, 128, 160, M2, CH, 1);
    gemm(Whc[4], Ha,   biasc[4], Hb, 128, 128, M2, CH, 1);
    gemm(Whc[5], Hb,   biasc[5], Hc, 256, 128, M2, CH, 1);
    pool_max<<<dim3((256 * 32 + 255) / 256, 1, CH), 256, 0, stream>>>(
        Hc, feat2 + (size_t)b0 * 256 * 32, 256, 32, 128, (long long)256 * 32, 32);
  }

  // ---- SA3 (group all) ----
  group_sa3<<<(64 * 32 + 255) / 256, 256, 0, stream>>>(nxyz2, feat2, Gbuf);
  gemm(Whc[6], Gbuf, biasc[6], Ha, 256, 288, 32, B, 1);
  gemm(Whc[7], Ha,   biasc[7], Hb, 512, 256, 32, B, 1);
  gemm(Whc[8], Hb,   biasc[8], Hc, 1024, 512, 32, B, 1);
  pool3_packed<<<dim3(2, 1, B), 256, 0, stream>>>(Hc, xTp);

  // ---- FC head (M = 64 batch columns) ----
  gemm(WhFc1, xTp, bFc1, Ha, 1024, 1024, 64, 1, 1);
  gemm(WhFc2, Ha,  bFc2, Hb, 1024, 1024, 64, 1, 1);
  fc3_sigmoid<<<B, 256, 0, stream>>>(Hb, Wh3, (const float*)d_in[38], (float*)d_out);
}
